// Transformer_85212151153299
// MI455X (gfx1250) — matrix operand order repaired and checked by reading, not compiled, not run on hardware
//
#include <hip/hip_runtime.h>
#include <hip/hip_bf16.h>
#include <stdint.h>

// ---------------------------------------------------------------------------
// Sizes
// ---------------------------------------------------------------------------
#define DIMV  129          // feature dim (2*64+1)
#define NC1   8193         // n+1 context columns
#define NCTX  8192
#define NCP   8208         // context padded to multiple of 16
#define DP    160          // feature padded to multiple of 32 (WMMA K)
#define DTR   144          // feature rows padded to multiple of 16 (9 tiles)
#define NT    513          // NCP / 16 tiles
#define LMBDF 0.9f
#define LAM128 1.3900845237714473e-6f   // 0.9^128
#define LCH   128          // scan chunk length
#define NCH   65           // ceil(8193/128)
#define INVN  (1.0f/8192.0f)
#define CPB   8            // columns per proj block

#define LZS   168          // padded LDS row stride for 160-elem rows (banks)
#define LGS   40           // padded LDS row stride for 32-elem G rows

typedef __attribute__((ext_vector_type(16))) __bf16 v16bf;
typedef __attribute__((ext_vector_type(8)))  __bf16 v8bf;
typedef __attribute__((ext_vector_type(8)))  float  v8f;
typedef __attribute__((ext_vector_type(4)))  int    v4i;

static __device__ __forceinline__ v8f v8f_zero() {
  v8f v = {0.f, 0.f, 0.f, 0.f, 0.f, 0.f, 0.f, 0.f};
  return v;
}

static __device__ __forceinline__ v8f wmma_bf16(v16bf a, v16bf b, v8f c) {
  // D = A(16x32 bf16) * B(32x16 bf16) + C(16x16 f32)
  return __builtin_amdgcn_wmma_f32_16x16x32_bf16(false, a, false, b, (short)0, c,
                                                 false, false);
}

// A-operand tile (16M x 32K bf16). Memory: row-major [m][k], stride in elems.
// Layout: lanes0-15 M=lane, K=0..15; lanes16-31 K=16..31.
static __device__ __forceinline__ v16bf load_a_tile(const __bf16* base, int row0,
                                                    int stride, int k0, int lane) {
  const __bf16* p = base + (size_t)(row0 + (lane & 15)) * stride + k0 +
                    ((lane & 16) ? 16 : 0);
  v8bf lo = *(const v8bf*)p;         // K = 0..7   (+16 in hi half)
  v8bf hi = *(const v8bf*)(p + 8);   // K = 8..15  (+16 in hi half)
  return __builtin_shufflevector(lo, hi, 0, 1, 2, 3, 4, 5, 6, 7, 8, 9, 10, 11,
                                 12, 13, 14, 15);
}

// B-operand tile (32K x 16N bf16). Memory: transposed row-major [n][k].
// Layout: lanes0-15 N=lane with K=0..15; lanes16-31 K=16..31.
static __device__ __forceinline__ v16bf load_b_tile(const __bf16* base, int col0,
                                                    int stride, int k0, int lane) {
  const __bf16* p = base + (size_t)(col0 + (lane & 15)) * stride + k0 +
                    ((lane & 16) ? 16 : 0);
  v8bf lo = *(const v8bf*)p;        // K = k0..k0+7
  v8bf hi = *(const v8bf*)(p + 8);  // K = k0+8..k0+15
  return __builtin_shufflevector(lo, hi, 0, 1, 2, 3, 4, 5, 6, 7, 8, 9, 10, 11,
                                 12, 13, 14, 15);
}

// ---------------------------------------------------------------------------
// CDNA5 async global->LDS staging (ASYNCcnt-tracked, no VGPR round trip).
// Signature (from compiler probe): (v4i AS1*, v4i AS3*, imm offset, imm cpol).
// Falls back to an explicit b128 copy if the builtins are unavailable.
// ---------------------------------------------------------------------------
#if __has_builtin(__builtin_amdgcn_global_load_async_to_lds_b128) && \
    __has_builtin(__builtin_amdgcn_s_wait_asynccnt)
#define USE_ASYNC_LDS 1
typedef __attribute__((address_space(1))) v4i* gas_v4i_ptr;  // global
typedef __attribute__((address_space(3))) v4i* las_v4i_ptr;  // LDS
#endif

static __device__ __forceinline__ void copy16_g2l(void* ldst, const void* gsrc) {
#ifdef USE_ASYNC_LDS
  __builtin_amdgcn_global_load_async_to_lds_b128((gas_v4i_ptr)gsrc,
                                                 (las_v4i_ptr)ldst, 0, 0);
#else
  *(uint4*)ldst = *(const uint4*)gsrc;
#endif
}

// Publish staged LDS data to the workgroup (waits the producer's ASYNCcnt,
// then barrier).  In the fallback path the compiler's dscnt wait + barrier
// provide the same guarantee.
static __device__ __forceinline__ void staging_fence() {
#ifdef USE_ASYNC_LDS
  __builtin_amdgcn_s_wait_asynccnt(0);
#endif
  __syncthreads();
}

// ---------------------------------------------------------------------------
// Workspace zero-fill (pads of Zt / Wt / Gb must be exactly 0)
// ---------------------------------------------------------------------------
__global__ void zero_kernel(uint32_t* __restrict__ p, size_t n32) {
  size_t i = (size_t)blockIdx.x * blockDim.x + threadIdx.x;
  size_t stride = (size_t)gridDim.x * blockDim.x;
  for (; i < n32; i += stride) p[i] = 0u;
}

// ---------------------------------------------------------------------------
// proj: W = Q*Z, PZ = P*Z (fp32, tiny GEMMs), also emit bf16 Z^T (Zt) and
// bf16 W^T (Wt), both feature-contiguous with zero padding to DP.
// ---------------------------------------------------------------------------
__global__ __launch_bounds__(288) void proj_kernel(
    const float* __restrict__ Z, const float* __restrict__ Qm,
    const float* __restrict__ Pm, __bf16* __restrict__ Zt,
    __bf16* __restrict__ Wt, float* __restrict__ PZ) {
  __shared__ float zl[CPB][DIMV + 3];
  int c0 = blockIdx.x * CPB;
  for (int idx = threadIdx.x; idx < CPB * DIMV; idx += 288) {
    int cc = idx / DIMV;
    int d = idx - cc * DIMV;
    int c = c0 + cc;
    zl[cc][d] = (c < NC1) ? Z[(size_t)d * NC1 + c] : 0.0f;
  }
  __syncthreads();
  int t = threadIdx.x;
  if (t < DIMV) {
    float acc[CPB];
#pragma unroll
    for (int cc = 0; cc < CPB; ++cc) acc[cc] = 0.f;
    const float* qrow = Qm + (size_t)t * DIMV;
    for (int i = 0; i < DIMV; ++i) {
      float qv = qrow[i];
#pragma unroll
      for (int cc = 0; cc < CPB; ++cc) acc[cc] = __fmaf_rn(qv, zl[cc][i], acc[cc]);
    }
#pragma unroll
    for (int cc = 0; cc < CPB; ++cc) {
      int c = c0 + cc;
      if (c < NC1) {
        Wt[(size_t)c * DP + t] = (__bf16)acc[cc];
        Zt[(size_t)c * DP + t] = (__bf16)zl[cc][t];
      }
    }
  } else if (t >= 144 && t < 144 + DIMV) {
    int d = t - 144;
    float acc[CPB];
#pragma unroll
    for (int cc = 0; cc < CPB; ++cc) acc[cc] = 0.f;
    const float* prow = Pm + (size_t)d * DIMV;
    for (int i = 0; i < DIMV; ++i) {
      float pv = prow[i];
#pragma unroll
      for (int cc = 0; cc < CPB; ++cc) acc[cc] = __fmaf_rn(pv, zl[cc][i], acc[cc]);
    }
#pragma unroll
    for (int cc = 0; cc < CPB; ++cc) {
      int c = c0 + cc;
      // Column n is multiplied by the zeroed last row of M -> force 0.
      if (c < NC1) PZ[(size_t)d * NCP + c] = (c == NCTX) ? 0.f : acc[cc];
    }
  }
}

// ---------------------------------------------------------------------------
// Decay-matrix multiply as a chunked suffix scan:
//   G[:,j] = PZ[:,j] + lambda * G[:,j+1],  G[:,n] = 0
// ---------------------------------------------------------------------------
__global__ __launch_bounds__(160) void scan_local_kernel(float* __restrict__ PZ) {
  int d = threadIdx.x;
  if (d >= DIMV) return;
  int start = blockIdx.x * LCH;
  int end = start + LCH;
  if (end > NC1) end = NC1;
  float* row = PZ + (size_t)d * NCP;
  float g = 0.f;
  for (int i = end - 1; i >= start; --i) {
    g = __fmaf_rn(LMBDF, g, row[i]);
    row[i] = g;
  }
}

__global__ __launch_bounds__(160) void scan_carry_kernel(
    const float* __restrict__ PZ, float* __restrict__ carry) {
  int d = threadIdx.x;
  if (d >= DIMV) return;
  const float* row = PZ + (size_t)d * NCP;
  float c = 0.f;  // true G value at the first column of the chunk to the right
  for (int ch = NCH - 1; ch >= 0; --ch) {
    carry[(size_t)d * NCH + ch] = c;
    int start = ch * LCH;
    int len = NC1 - start;
    if (len > LCH) len = LCH;
    float lp = (len == LCH) ? LAM128 : LMBDF;  // only last chunk has len 1
    c = __fmaf_rn(lp, c, row[start]);
  }
}

__global__ __launch_bounds__(160) void scan_apply_kernel(
    const float* __restrict__ PZ, const float* __restrict__ carry,
    __bf16* __restrict__ Gb) {
  int d = threadIdx.x;
  if (d >= DIMV) return;
  int start = blockIdx.x * LCH;
  int end = start + LCH;
  if (end > NC1) end = NC1;
  float cr = carry[(size_t)d * NCH + blockIdx.x];
  const float* row = PZ + (size_t)d * NCP;
  __bf16* grow = Gb + (size_t)d * NCP;
  float f = LMBDF;
  for (int i = end - 1; i >= start; --i) {
    grow[i] = (__bf16)__fmaf_rn(f, cr, row[i]);
    f *= LMBDF;
  }
}

// ---------------------------------------------------------------------------
// Phase 1: row sums of exp(X), X = Zt * Wt^T.  softmax is shift-invariant and
// |X| = O(1) here (Q std ~0.009), so no max pass is needed.  Padded columns
// produce X = 0 exactly (zero-padded Wt), i.e. exp contributes exactly 1.0,
// so the loop runs unmasked and the constant is subtracted once at the end.
// 4 waves/block, each owning a 16-row k-tile with resident Zt A-operands;
// Wt j-slabs stream through double-buffered LDS via async global->LDS.
// ---------------------------------------------------------------------------
__global__ __launch_bounds__(128) void stats_kernel(
    const __bf16* __restrict__ Zt, const __bf16* __restrict__ Wt,
    float* __restrict__ rsArr) {
  __shared__ __bf16 lW[2][16 * LZS];
  int lane = threadIdx.x & 31;
  int kt = blockIdx.x * 4 + (threadIdx.x >> 5);
  bool active = kt < NT;
  int kbase = active ? kt * 16 : 0;

  v16bf A[5];
  if (active) {
#pragma unroll
    for (int c = 0; c < 5; ++c) A[c] = load_a_tile(Zt, kbase, DP, c * 32, lane);
  }
  float sloc[8];
#pragma unroll
  for (int r = 0; r < 8; ++r) sloc[r] = 0.f;

  // stage j-tile 0 (16 rows x 20 b128 chunks = 320)
#pragma unroll
  for (int u = 0; u < 3; ++u) {
    int c = threadIdx.x + u * 128;
    if (c < 320) {
      int row = c / 20, ch = c - row * 20;
      copy16_g2l(&lW[0][row * LZS + ch * 8], Wt + (size_t)row * DP + ch * 8);
    }
  }
  staging_fence();

  for (int jt = 0; jt < NT; ++jt) {
    int cur = jt & 1;
    if (jt + 1 < NT) {  // stage next slab while current is consumed
      const __bf16* src = Wt + (size_t)(jt + 1) * 16 * DP;
      __bf16* dst = lW[cur ^ 1];
#pragma unroll
      for (int u = 0; u < 3; ++u) {
        int c = threadIdx.x + u * 128;
        if (c < 320) {
          int row = c / 20, ch = c - row * 20;
          copy16_g2l(&dst[row * LZS + ch * 8], src + (size_t)row * DP + ch * 8);
        }
      }
    }
    if (active) {
      v16bf Bt[5];  // batch ds_loads so one wait covers all 5 WMMAs
#pragma unroll
      for (int c = 0; c < 5; ++c)
        Bt[c] = load_b_tile(lW[cur], 0, LZS, c * 32, lane);
      v8f X = v8f_zero();
#pragma unroll
      for (int c = 0; c < 5; ++c) X = wmma_bf16(A[c], Bt[c], X);
#pragma unroll
      for (int r = 0; r < 8; ++r) sloc[r] += __expf(X[r]);
    }
    staging_fence();
  }

  if (active) {
    // remove the exp(0)=1 contribution of the padded columns this lane saw
    float corr = ((lane & 15) != 0) ? 1.f : 0.f;
#pragma unroll
    for (int r = 0; r < 8; ++r) sloc[r] -= corr;
#pragma unroll
    for (int off = 1; off < 16; off <<= 1) {
#pragma unroll
      for (int r = 0; r < 8; ++r) sloc[r] += __shfl_xor(sloc[r], off, 32);
    }
    if ((lane & 15) == 0) {
#pragma unroll
      for (int r = 0; r < 8; ++r)
        rsArr[kbase + r + ((lane & 16) ? 8 : 0)] = 1.0f / sloc[r];
    }
  }
}

// ---------------------------------------------------------------------------
// Phase 2: Znew = Zin + (1/n) * G * softmax(X).  4 waves/block each own a
// 16-column j-strip with resident Wt B-operands; Zt (32 k-rows) and Gb
// (144 x 32) k-slabs stream through double-buffered LDS via async
// global->LDS.  exp(X) tiles repack into the B operand of the G-GEMM with a
// single half-wave shfl_xor(16).
// ---------------------------------------------------------------------------
static __device__ __forceinline__ void stage_pass2(const __bf16* __restrict__ Zt,
                                                   const __bf16* __restrict__ Gb,
                                                   __bf16* zdst, __bf16* gdst,
                                                   int kb, int tid) {
  const __bf16* zsrc = Zt + (size_t)kb * DP;
  const __bf16* gsrc = Gb + kb;
  // 640 chunks of Zt slab (32 rows x 20) + 576 chunks of Gb slab (144 x 4)
#pragma unroll
  for (int u = 0; u < 10; ++u) {
    int c = tid + u * 128;
    if (c < 640) {
      int row = c / 20, ch = c - row * 20;
      copy16_g2l(&zdst[row * LZS + ch * 8], zsrc + (size_t)row * DP + ch * 8);
    } else if (c < 1216) {
      int cc = c - 640;
      int row = cc >> 2, ch = cc & 3;
      copy16_g2l(&gdst[row * LGS + ch * 8], gsrc + (size_t)row * NCP + ch * 8);
    }
  }
}

__global__ __launch_bounds__(128) void pass2_kernel(
    const __bf16* __restrict__ Zt, const __bf16* __restrict__ Wt,
    const __bf16* __restrict__ Gb, const float* __restrict__ rsArr,
    const float* __restrict__ Zin, float* __restrict__ Zout) {
  __shared__ __bf16 lZ[2][32 * LZS];
  __shared__ __bf16 lG[2][DTR * LGS];
  int lane = threadIdx.x & 31;
  int jt = blockIdx.x * 4 + (threadIdx.x >> 5);
  bool active = jt < NT;
  int j0 = active ? jt * 16 : 0;
  int hi = (lane & 16) ? 1 : 0;

  v16bf Bw[5];  // resident Wt B-operands for this j-strip (K = 160)
  if (active) {
#pragma unroll
    for (int c = 0; c < 5; ++c) Bw[c] = load_b_tile(Wt, j0, DP, c * 32, lane);
  }
  v8f acc[9];
#pragma unroll
  for (int dt = 0; dt < 9; ++dt) acc[dt] = v8f_zero();

  stage_pass2(Zt, Gb, lZ[0], lG[0], 0, threadIdx.x);
  staging_fence();

  for (int kb = 0; kb < NCP; kb += 32) {
    int cur = (kb >> 5) & 1;
    if (kb + 32 < NCP)
      stage_pass2(Zt, Gb, lZ[cur ^ 1], lG[cur ^ 1], kb + 32, threadIdx.x);
    if (active) {
      // X tiles for k = kb..kb+15 (c0) and kb+16..kb+31 (c1), from LDS.
      // Batch the 5 A-tiles per row group so one ds wait covers 5 WMMAs;
      // the array is reused for the second group to cap register pressure.
      v8f c0 = v8f_zero(), c1 = v8f_zero();
      v16bf za[5];
#pragma unroll
      for (int c = 0; c < 5; ++c) za[c] = load_a_tile(lZ[cur], 0, LZS, c * 32, lane);
#pragma unroll
      for (int c = 0; c < 5; ++c) c0 = wmma_bf16(za[c], Bw[c], c0);
#pragma unroll
      for (int c = 0; c < 5; ++c) za[c] = load_a_tile(lZ[cur], 16, LZS, c * 32, lane);
#pragma unroll
      for (int c = 0; c < 5; ++c) c1 = wmma_bf16(za[c], Bw[c], c1);

      // normalize (no shift needed; see stats_kernel) and repack as B operand
      v8f rlo = *(const v8f*)(rsArr + kb + hi * 8);       // rows of c0
      v8f rhi = *(const v8f*)(rsArr + kb + 16 + hi * 8);  // rows of c1
      v16bf Bp;
#pragma unroll
      for (int r = 0; r < 8; ++r) {
        float o0 = __expf(c0[r]) * rlo[r];
        float o1 = __expf(c1[r]) * rhi[r];
        float q0 = __shfl_xor(o0, 16, 32);  // partner half's rows
        float q1 = __shfl_xor(o1, 16, 32);
        // B layout: lo lanes hold K=0..15 (all from c0), hi lanes K=16..31 (c1)
        Bp[r] = (__bf16)(hi ? q1 : o0);
        Bp[r + 8] = (__bf16)(hi ? o1 : q0);
      }
      // out accumulation: 9 feature tiles, A-operand from staged Gb slab
      v16bf g = load_a_tile(lG[cur], 0, LGS, 0, lane);
#pragma unroll
      for (int dt = 0; dt < 9; ++dt) {
        int ndt = (dt + 1 < 9) ? dt + 1 : 8;
        v16bf gn = load_a_tile(lG[cur], ndt * 16, LGS, 0, lane);
        acc[dt] = wmma_bf16(g, Bp, acc[dt]);
        g = gn;
      }
    }
    staging_fence();
  }

  // residual add + store (guard real dims)
  int j = j0 + (lane & 15);
  if (active && j < NC1) {
#pragma unroll
    for (int dt = 0; dt < 9; ++dt) {
#pragma unroll
      for (int r = 0; r < 8; ++r) {
        int d = dt * 16 + r + hi * 8;
        if (d < DIMV) {
          size_t idx = (size_t)d * NC1 + j;
          Zout[idx] = __fmaf_rn(INVN, acc[dt][r], Zin[idx]);
        }
      }
    }
  }
}

// ---------------------------------------------------------------------------
// Host-side launcher
// ---------------------------------------------------------------------------
extern "C" void kernel_launch(void* const* d_in, const int* in_sizes, int n_in,
                              void* d_out, int out_size, void* d_ws,
                              size_t ws_size, hipStream_t stream) {
  (void)in_sizes; (void)n_in; (void)out_size; (void)ws_size;
  const float* Z0 = (const float*)d_in[0];
  const float* Pm = (const float*)d_in[1];
  const float* Qm = (const float*)d_in[2];
  float* Zfinal = (float*)d_out;

  uint8_t* ws = (uint8_t*)d_ws;
  size_t off = 0;
  auto take = [&](size_t bytes) -> uint8_t* {
    uint8_t* p = ws + off;
    off += (bytes + 255) & ~(size_t)255;
    return p;
  };
  __bf16* Zt = (__bf16*)take((size_t)NCP * DP * 2);   // Z^T  bf16 [8208][160]
  __bf16* Wt = (__bf16*)take((size_t)NCP * DP * 2);   // (QZ)^T bf16
  __bf16* Gb = (__bf16*)take((size_t)DTR * NCP * 2);  // G bf16 [144][8208]
  size_t zero_bytes = off;                            // pads must be 0
  float* PZ = (float*)take((size_t)DIMV * NCP * 4);   // P*Z / local scan
  float* carry = (float*)take((size_t)DIMV * NCH * 4);
  float* rsArr = (float*)take((size_t)NCP * 4);
  float* ZA = (float*)take((size_t)DIMV * NC1 * 4);
  float* ZB = (float*)take((size_t)DIMV * NC1 * 4);

  zero_kernel<<<2048, 256, 0, stream>>>((uint32_t*)ws, zero_bytes / 4);

  const float* zin = Z0;
  for (int layer = 0; layer < 4; ++layer) {
    float* zout = (layer == 3) ? Zfinal : ((layer & 1) ? ZB : ZA);
    proj_kernel<<<(NC1 + CPB - 1) / CPB, 288, 0, stream>>>(zin, Qm, Pm, Zt, Wt,
                                                           PZ);
    scan_local_kernel<<<NCH, 160, 0, stream>>>(PZ);
    scan_carry_kernel<<<1, 160, 0, stream>>>(PZ, carry);
    scan_apply_kernel<<<NCH, 160, 0, stream>>>(PZ, carry, Gb);
    stats_kernel<<<(NT + 3) / 4, 128, 0, stream>>>(Zt, Wt, rsArr);
    pass2_kernel<<<(NT + 3) / 4, 128, 0, stream>>>(Zt, Wt, Gb, rsArr, zin, zout);
    zin = zout;
  }
}
